// BondLengthHead_48009144434794
// MI455X (gfx1250) — matrix operand order repaired and checked by reading, not compiled, not run on hardware
//
#include <hip/hip_runtime.h>
#include <hip/hip_bf16.h>

#define WAVES 8
#define TPB (WAVES * 32)

typedef __attribute__((ext_vector_type(16))) __bf16 v16bf;
typedef __attribute__((ext_vector_type(8)))  float  v8f;

union Frag16 { v16bf v; unsigned int u[8]; uint4 q[2]; };
union FragC  { v8f   v; float f[8]; };

// One-time (cold path) RNE pack for weights / z pre-conversion.
__device__ __forceinline__ unsigned int pack_bf16_rne(float a, float b) {
    unsigned int ua = __float_as_uint(a);
    unsigned int ub = __float_as_uint(b);
    ua += 0x7FFFu + ((ua >> 16) & 1u);
    ub += 0x7FFFu + ((ub >> 16) & 1u);
    return (ua >> 16) | (ub & 0xFFFF0000u);
}
// Hot-path pack: single v_perm_b32 (truncate both f32 to bf16, pack lo/hi).
__device__ __forceinline__ unsigned int pack_bf16_trunc(float a, float b) {
    return __builtin_amdgcn_perm(__float_as_uint(b), __float_as_uint(a), 0x07060302u);
}

// f32 -> packed bf16 pre-conversion of z / embed into workspace.
__global__ void cvt_f32_to_bf16_kernel(const float* __restrict__ src,
                                       unsigned int* __restrict__ dst, int npairs) {
    int i = blockIdx.x * blockDim.x + threadIdx.x;
    if (i < npairs) {
        float2 f = ((const float2*)src)[i];
        dst[i] = pack_bf16_rne(f.x, f.y);
    }
}

// MLP over gathered edge features, bf16 WMMA with f32 accumulate.
// Layer1: [16x288]@[288x128], Layer2: [16x128]@[128x64], Layer3: VALU dot(64)+softplus.
// PRE=true: z/embed already bf16 in workspace -> A fragments are raw 16B loads.
template <bool PRE>
__global__ __launch_bounds__(TPB)
void bond_mlp_wmma_kernel(const float* __restrict__ z,
                          const int*   __restrict__ edge_index,
                          const int*   __restrict__ bond_types,
                          const float* __restrict__ embed,
                          const float* __restrict__ W1, const float* __restrict__ b1,
                          const float* __restrict__ W2, const float* __restrict__ b2,
                          const float* __restrict__ W3, const float* __restrict__ b3,
                          const unsigned short* __restrict__ zb,
                          const unsigned short* __restrict__ eb,
                          float* __restrict__ out, int E, int ntiles)
{
    // W1 as 9(Kblk) x 8(Ntile) B-fragments, each 32 lanes x 8 dwords of packed bf16 pairs.
    __shared__ __align__(16) unsigned int   sW1[9 * 8 * 32 * 8];   // 72 KB
    __shared__ __align__(16) unsigned int   sW2[4 * 4 * 32 * 8];   // 16 KB
    __shared__ __align__(16) float          sW3[64];
    __shared__ float                        sB1[128];
    __shared__ float                        sB2[64];
    __shared__ float                        sB3v[1];
    __shared__ __align__(16) unsigned short sH[WAVES][16 * 128];   // per-wave scratch, 32 KB

    const int tid = threadIdx.x;

    // ---- Stage weights into LDS, pre-swizzled into WMMA B-fragment layout ----
    // B layout (32xK x 16xN per frag): lanes 0-15 hold K {0..7,16..23}, lanes 16-31 hold
    // K {8..15,24..31} (dword j = K pair); lane % 16 selects column N.
    for (int d = tid; d < 9 * 8 * 32 * 8; d += TPB) {
        int j    = d & 7;
        int lane = (d >> 3) & 31;
        int frag = d >> 8;
        int nt = frag & 7, kb = frag >> 3;
        int half = lane >> 4, nl = lane & 15;
        int k0 = kb * 32 + half * 8 + (j & 4) * 2 + j * 2;
        int n  = nt * 16 + nl;
        sW1[d] = pack_bf16_rne(W1[k0 * 128 + n], W1[(k0 + 1) * 128 + n]);
    }
    for (int d = tid; d < 4 * 4 * 32 * 8; d += TPB) {
        int j    = d & 7;
        int lane = (d >> 3) & 31;
        int frag = d >> 8;
        int nt = frag & 3, kb = frag >> 2;
        int half = lane >> 4, nl = lane & 15;
        int k0 = kb * 32 + half * 8 + (j & 4) * 2 + j * 2;
        int n  = nt * 16 + nl;
        sW2[d] = pack_bf16_rne(W2[k0 * 64 + n], W2[(k0 + 1) * 64 + n]);
    }
    if (tid < 64)  sW3[tid] = W3[tid];
    if (tid < 128) sB1[tid] = b1[tid];
    if (tid < 64)  sB2[tid] = b2[tid];
    if (tid == 0)  sB3v[0] = b3[0];
    __syncthreads();

    const int wid  = tid >> 5;
    const int lane = tid & 31;
    const int ml   = lane & 15;   // M (edge within tile) this lane serves
    const int kh   = lane >> 4;   // K-half selector for A layout

    unsigned short* h1 = sH[wid];
    float*          h2 = (float*)sH[wid];

    const int tstride = gridDim.x * WAVES;
    const v8f zeroC = {};        // folds to inline SRC2 = 0 on the first WMMA

    for (int tile = blockIdx.x * WAVES + wid; tile < ntiles; tile += tstride) {
        int e  = tile * 16 + ml;
        int ec = (e < E) ? e : 0;
        int src = edge_index[ec];
        int dst = edge_index[E + ec];
        int bt  = bond_types[ec];

        // Pull next tile's index data toward the caches (global_prefetch_b8).
        int epf = e + tstride * 16;
        if (epf < E) {
            __builtin_prefetch(&edge_index[epf], 0, 0);
            __builtin_prefetch(&edge_index[E + epf], 0, 0);
            __builtin_prefetch(&bond_types[epf], 0, 0);
        }

        // A-fragment loader. A layout: lane<16 needs K {0..7,16..23} of the
        // 32-wide block, lane>=16 needs {8..15,24..31}.
        auto loadA = [&](int kb) {
            Frag16 A;
            if constexpr (PRE) {
                const unsigned short* p = (kb < 4) ? (zb + (size_t)src * 128 + kb * 32)
                                        : (kb < 8) ? (zb + (size_t)dst * 128 + (kb - 4) * 32)
                                                   : (eb + (size_t)bt * 32);
                const uint4* pv = (const uint4*)(p + kh * 8);
                A.q[0] = pv[0];
                A.q[1] = pv[2];          // +16 bf16 elements = +32 bytes
            } else {
                const float* p = (kb < 4) ? (z + (size_t)src * 128 + kb * 32)
                               : (kb < 8) ? (z + (size_t)dst * 128 + (kb - 4) * 32)
                                          : (embed + (size_t)bt * 32);
                const float4* pv = (const float4*)(p + kh * 8);
                float4 a0 = pv[0], a1 = pv[1], a2 = pv[4], a3 = pv[5];
                A.u[0] = pack_bf16_trunc(a0.x, a0.y); A.u[1] = pack_bf16_trunc(a0.z, a0.w);
                A.u[2] = pack_bf16_trunc(a1.x, a1.y); A.u[3] = pack_bf16_trunc(a1.z, a1.w);
                A.u[4] = pack_bf16_trunc(a2.x, a2.y); A.u[5] = pack_bf16_trunc(a2.z, a2.w);
                A.u[6] = pack_bf16_trunc(a3.x, a3.y); A.u[7] = pack_bf16_trunc(a3.z, a3.w);
            }
            return A;
        };
        auto loadB1 = [&](int kb, int nt) {
            Frag16 B;
            const uint4* bp = (const uint4*)&sW1[((kb * 8 + nt) * 32 + lane) * 8];
            B.q[0] = bp[0]; B.q[1] = bp[1];
            return B;
        };

        // ---- Layer 1: features[16x288] @ W1[288x128] ----
        FragC acc[8];
        {   // kb = 0 peeled: C is literal zero -> no accumulator init movs.
            Frag16 A = loadA(0);
            Frag16 B[8];
#pragma unroll
            for (int nt = 0; nt < 8; nt++) B[nt] = loadB1(0, nt);
#pragma unroll
            for (int nt = 0; nt < 8; nt++)
                acc[nt].v = __builtin_amdgcn_wmma_f32_16x16x32_bf16(
                    false, A.v, false, B[nt].v, (short)0, zeroC, false, false);
        }
        for (int kb = 1; kb < 9; kb++) {
            Frag16 A = loadA(kb);
            Frag16 B[8];
#pragma unroll
            for (int nt = 0; nt < 8; nt++) B[nt] = loadB1(kb, nt);
#pragma unroll
            for (int nt = 0; nt < 8; nt++)
                acc[nt].v = __builtin_amdgcn_wmma_f32_16x16x32_bf16(
                    false, A.v, false, B[nt].v, (short)0, acc[nt].v, false, false);
        }

        // bias + relu -> h1 (bf16, row-major 16x128 in per-wave LDS); pairwise
        // v_perm pack, then lo/hi 16-bit stores (rows m, m+1 share one dword).
#pragma unroll
        for (int nt = 0; nt < 8; nt++) {
            int n = nt * 16 + ml;
            float bias = sB1[n];
#pragma unroll
            for (int r = 0; r < 8; r += 2) {
                int m = r + kh * 8;                      // C layout: lane half selects M+8
                float v0 = fmaxf(acc[nt].f[r]     + bias, 0.f);
                float v1 = fmaxf(acc[nt].f[r + 1] + bias, 0.f);
                unsigned int pk = pack_bf16_trunc(v0, v1);
                h1[m * 128 + n]       = (unsigned short)pk;
                h1[(m + 1) * 128 + n] = (unsigned short)(pk >> 16);
            }
        }

        // ---- Layer 2: h1[16x128] @ W2[128x64] ----
        auto loadA2 = [&](int kb) {
            Frag16 A2;
            const uint4* ap0 = (const uint4*)&h1[ml * 128 + kb * 32 + kh * 8];
            const uint4* ap1 = (const uint4*)&h1[ml * 128 + kb * 32 + kh * 8 + 16];
            A2.q[0] = ap0[0]; A2.q[1] = ap1[0];
            return A2;
        };
        auto loadB2 = [&](int kb, int nt) {
            Frag16 B;
            const uint4* bp = (const uint4*)&sW2[((kb * 4 + nt) * 32 + lane) * 8];
            B.q[0] = bp[0]; B.q[1] = bp[1];
            return B;
        };

        FragC acc2[4];
        {   // kb = 0 peeled with inline-zero C.
            Frag16 A2 = loadA2(0);
            Frag16 B[4];
#pragma unroll
            for (int nt = 0; nt < 4; nt++) B[nt] = loadB2(0, nt);
#pragma unroll
            for (int nt = 0; nt < 4; nt++)
                acc2[nt].v = __builtin_amdgcn_wmma_f32_16x16x32_bf16(
                    false, A2.v, false, B[nt].v, (short)0, zeroC, false, false);
        }
#pragma unroll
        for (int kb = 1; kb < 4; kb++) {
            Frag16 A2 = loadA2(kb);
            Frag16 B[4];
#pragma unroll
            for (int nt = 0; nt < 4; nt++) B[nt] = loadB2(kb, nt);
#pragma unroll
            for (int nt = 0; nt < 4; nt++)
                acc2[nt].v = __builtin_amdgcn_wmma_f32_16x16x32_bf16(
                    false, A2.v, false, B[nt].v, (short)0, acc2[nt].v, false, false);
        }

        // bias + relu -> h2 (f32 16x64, reuses per-wave LDS; DS in-order per wave)
#pragma unroll
        for (int nt = 0; nt < 4; nt++) {
            int n = nt * 16 + ml;
            float bias = sB2[n];
#pragma unroll
            for (int r = 0; r < 8; r++) {
                int m = r + kh * 8;
                h2[m * 64 + n] = fmaxf(acc2[nt].f[r] + bias, 0.f);
            }
        }

        // ---- Layer 3: out = softplus(h2 . W3 + b3), one edge per lane (lanes 0..15) ----
        if (lane < 16) {
            const float* row = h2 + ml * 64;
            float s = sB3v[0];
#pragma unroll
            for (int i = 0; i < 64; i += 4) {
                float4 hv = *(const float4*)(row + i);
                float4 wv = *(const float4*)(sW3 + i);
                s += hv.x * wv.x + hv.y * wv.y + hv.z * wv.z + hv.w * wv.w;
            }
            float y = (s > 20.f) ? s : log1pf(__expf(s));
            if (e < E) out[e] = y;
        }
    }
}

extern "C" void kernel_launch(void* const* d_in, const int* in_sizes, int n_in,
                              void* d_out, int out_size, void* d_ws, size_t ws_size,
                              hipStream_t stream) {
    const float* z          = (const float*)d_in[0];
    const int*   edge_index = (const int*)d_in[1];
    const int*   bond_types = (const int*)d_in[2];
    const float* embed      = (const float*)d_in[3];
    const float* W1         = (const float*)d_in[4];
    const float* b1         = (const float*)d_in[5];
    const float* W2         = (const float*)d_in[6];
    const float* b2         = (const float*)d_in[7];
    const float* W3         = (const float*)d_in[8];
    const float* b3         = (const float*)d_in[9];
    float* out = (float*)d_out;

    int E = in_sizes[1] / 2;                 // edge_index is [2, E]
    int ntiles = (E + 15) / 16;
    int blocks = (ntiles + WAVES - 1) / WAVES;
    if (blocks > 1024) blocks = 1024;
    if (blocks < 1) blocks = 1;

    size_t zbytes = (size_t)in_sizes[0] * 2;                 // z as bf16
    size_t eoff   = (zbytes + 15) & ~(size_t)15;
    size_t need   = eoff + (size_t)in_sizes[3] * 2;          // + embed as bf16

    if (ws_size >= need) {
        unsigned short* zb = (unsigned short*)d_ws;
        unsigned short* ebv = (unsigned short*)((char*)d_ws + eoff);
        int npz = in_sizes[0] / 2;
        int npe = in_sizes[3] / 2;
        cvt_f32_to_bf16_kernel<<<(npz + 255) / 256, 256, 0, stream>>>(z, (unsigned int*)zb, npz);
        cvt_f32_to_bf16_kernel<<<(npe + 255) / 256, 256, 0, stream>>>(embed, (unsigned int*)ebv, npe);
        bond_mlp_wmma_kernel<true><<<blocks, TPB, 0, stream>>>(
            z, edge_index, bond_types, embed, W1, b1, W2, b2, W3, b3,
            zb, ebv, out, E, ntiles);
    } else {
        bond_mlp_wmma_kernel<false><<<blocks, TPB, 0, stream>>>(
            z, edge_index, bond_types, embed, W1, b1, W2, b2, W3, b3,
            nullptr, nullptr, out, E, ntiles);
    }
}
